// DecoderLayer_58695023068058
// MI455X (gfx1250) — matrix operand order repaired and checked by reading, not compiled, not run on hardware
//
#include <hip/hip_runtime.h>
#include <hip/hip_bf16.h>

// ---------------------------------------------------------------------------
// Transformer decoder layer for MI455X (gfx1250, wave32, WMMA).
// B=4, S=T=2048, DIN=256, DM=512, H=8, DH=64.
// - One-time f32->bf16 conversion passes (weights pre-transposed to [N][K]).
// - All GEMM-shaped math via v_wmma_f32_16x16x32_bf16 (bf16 in, f32 acc).
// - Flash attention (online softmax), scores never materialized.
// - Tile staging via GLOBAL_LOAD_ASYNC_TO_LDS_B128 (ASYNCcnt) when available.
// ---------------------------------------------------------------------------

typedef __attribute__((ext_vector_type(16))) __bf16 v16bf;
typedef __attribute__((ext_vector_type(8)))  float  v8f;

#define WMMA_BF16(A, B, C) \
  __builtin_amdgcn_wmma_f32_16x16x32_bf16(false, (A), false, (B), (short)0, (C), false, false)

#if defined(__has_builtin)
#  if __has_builtin(__builtin_amdgcn_global_load_async_to_lds_b128)
#    define USE_ASYNC_LDS 1
#  endif
#endif
#ifndef USE_ASYNC_LDS
#  define USE_ASYNC_LDS 0
#endif

__device__ __forceinline__ unsigned short f2bf(float f) {
  unsigned int u = __float_as_uint(f);
  unsigned int r = u + 0x7FFFu + ((u >> 16) & 1u);   // round to nearest even
  return (unsigned short)(r >> 16);
}

// Copy 16 bytes global -> LDS. Async (ASYNCcnt) path on CDNA5 toolchains.
__device__ __forceinline__ void stage16(unsigned short* lds_dst,
                                        const unsigned short* g_src) {
#if USE_ASYNC_LDS
  typedef int v4i __attribute__((vector_size(16)));
  // Builtin signature: (v4i* global_src, v4i* lds_dst, imm offset, imm cpol).
  // Pass generic v4i*; clang rewrites pointer address spaces for builtins.
  __builtin_amdgcn_global_load_async_to_lds_b128((v4i*)g_src, (v4i*)lds_dst, 0, 0);
#else
  *(uint4*)lds_dst = *(const uint4*)g_src;
#endif
}

__device__ __forceinline__ void stage_wait() {
#if USE_ASYNC_LDS
#  if __has_builtin(__builtin_amdgcn_s_wait_asynccnt)
  __builtin_amdgcn_s_wait_asynccnt(0);
#  else
  asm volatile("s_wait_asynccnt 0x0" ::: "memory");
#  endif
#endif
}

// Load a 16-element bf16 fragment as two contiguous 8-halfword (16B) LDS reads.
__device__ __forceinline__ v16bf ld_frag(const unsigned short* p0,
                                         const unsigned short* p1) {
  union { uint4 q[2]; v16bf v; } u;
  u.q[0] = *(const uint4*)p0;
  u.q[1] = *(const uint4*)p1;
  return u.v;
}

__device__ __forceinline__ float rmax16(float v) {
  v = fmaxf(v, __shfl_xor(v, 1, 32));
  v = fmaxf(v, __shfl_xor(v, 2, 32));
  v = fmaxf(v, __shfl_xor(v, 4, 32));
  v = fmaxf(v, __shfl_xor(v, 8, 32));
  return v;
}
__device__ __forceinline__ float rsum16(float v) {
  v += __shfl_xor(v, 1, 32);
  v += __shfl_xor(v, 2, 32);
  v += __shfl_xor(v, 4, 32);
  v += __shfl_xor(v, 8, 32);
  return v;
}

// ---------------------------------------------------------------------------
// Elementwise f32 -> bf16 (4 per thread).
// ---------------------------------------------------------------------------
__global__ __launch_bounds__(256) void cvt_kernel(
    const float* __restrict__ in, unsigned short* __restrict__ out, int n4)
{
  int i = blockIdx.x * 256 + threadIdx.x;
  if (i >= n4) return;
  float4 f = ((const float4*)in)[i];
  uint2 p;
  p.x = (unsigned int)f2bf(f.x) | ((unsigned int)f2bf(f.y) << 16);
  p.y = (unsigned int)f2bf(f.z) | ((unsigned int)f2bf(f.w) << 16);
  ((uint2*)out)[i] = p;
}

// W[K][N] f32  ->  Wt[N][K] bf16 (transpose while converting).
__global__ __launch_bounds__(256) void cvtT_kernel(
    const float* __restrict__ in, unsigned short* __restrict__ out, int K, int N)
{
  int i = blockIdx.x * 256 + threadIdx.x;   // i over K*N, coalesced on N
  if (i >= K * N) return;
  int k = i / N, n = i - k * N;
  out[(size_t)n * K + k] = f2bf(in[i]);
}

// ---------------------------------------------------------------------------
// GEMM: C[M,N] = A[M,K](bf16) @ Wt[N,K](bf16)^T + bias (+res) (+ReLU)
// 64x64 tile / 128 threads (4 waves); each wave: 16 rows x 64 cols.
// ---------------------------------------------------------------------------
__global__ __launch_bounds__(128) void gemm_bf16_kernel(
    const unsigned short* __restrict__ A,   // [M][K] bf16
    const unsigned short* __restrict__ Wt,  // [N][K] bf16 (pre-transposed)
    const float* __restrict__ bias, const float* __restrict__ res,
    void* __restrict__ Cout, int M, int N, int K, int relu, int out_bf16)
{
  __shared__ alignas(16) unsigned short sA[64 * 32];  // [m][k]
  __shared__ alignas(16) unsigned short sB[64 * 32];  // [n][k]

  const int tid  = threadIdx.x;
  const int lane = tid & 31;
  const int wave = tid >> 5;
  const int ln   = lane & 15;
  const int kh   = lane >> 4;
  const int mBase = blockIdx.y * 64;
  const int nBase = blockIdx.x * 64;

  const v8f vzero = {};
  v8f acc[4];
#pragma unroll
  for (int i = 0; i < 4; ++i) acc[i] = vzero;

  for (int kb = 0; kb < K; kb += 32) {
    __syncthreads();
#pragma unroll
    for (int t = 0; t < 2; ++t) {           // 2048 bf16 per tile, 16B granules
      int e = (tid + t * 128) * 8;
      int r = e >> 5, c = e & 31;
      stage16(&sA[e], &A [(size_t)(mBase + r) * K + kb + c]);
      stage16(&sB[e], &Wt[(size_t)(nBase + r) * K + kb + c]);
    }
    stage_wait();
    __syncthreads();

    const int row = wave * 16 + ln;
    v16bf a = ld_frag(&sA[row * 32 + kh * 16],
                      &sA[row * 32 + kh * 16 + 8]);
#pragma unroll
    for (int jn = 0; jn < 4; ++jn) {
      const int nl = jn * 16 + ln;
      v16bf b = ld_frag(&sB[nl * 32 + kh * 16],
                        &sB[nl * 32 + kh * 16 + 8]);
      acc[jn] = WMMA_BF16(a, b, acc[jn]);
    }
  }

  const int mRow0 = mBase + wave * 16 + 8 * kh;
#pragma unroll
  for (int jn = 0; jn < 4; ++jn) {
    const int col = nBase + jn * 16 + ln;
    const float bcol = bias ? bias[col] : 0.0f;
#pragma unroll
    for (int r = 0; r < 8; ++r) {
      size_t o = (size_t)(mRow0 + r) * N + col;
      float v = acc[jn][r] + bcol;
      if (res)  v += res[o];
      if (relu) v = fmaxf(v, 0.0f);
      if (out_bf16) ((unsigned short*)Cout)[o] = f2bf(v);
      else          ((float*)Cout)[o] = v;
    }
  }
}

// ---------------------------------------------------------------------------
// Flash attention: block = (64 q-rows, one head, one batch); wave = 16 q-rows.
// Q/K/V/O are bf16 [rows][512] with head slice h*64..h*64+63. DH=64.
// ---------------------------------------------------------------------------
#define NEGF (-4294967295.0f)

__global__ __launch_bounds__(128) void attn_kernel(
    const unsigned short* __restrict__ Qg, const unsigned short* __restrict__ Kg,
    const unsigned short* __restrict__ Vg, const int* __restrict__ mask,
    unsigned short* __restrict__ Og, int Sq, int Sk, int masked)
{
  __shared__ alignas(16) unsigned short sQ[64 * 64];      // [q][d]
  __shared__ alignas(16) unsigned short sK[32 * 64];      // [key][d]
  __shared__ alignas(16) unsigned short sV[64 * 32];      // [d][key]
  __shared__ alignas(16) unsigned short sP[4 * 16 * 32];  // per-wave P scratch

  const int tid  = threadIdx.x;
  const int lane = tid & 31;
  const int wave = tid >> 5;
  const int ln   = lane & 15;
  const int kh   = lane >> 4;
  const int b  = blockIdx.z;
  const int h  = blockIdx.y;
  const int qt = blockIdx.x;

  const size_t qrow0 = (size_t)b * Sq + qt * 64;
  const size_t krow0 = (size_t)b * Sk;
  const size_t hoff  = (size_t)h * 64;

  // Stage Q once (raw copy; 1/sqrt(DH) folded into softmax below).
#pragma unroll
  for (int t = 0; t < 4; ++t) {
    int e = (tid + t * 128) * 8;             // 4096 elements
    int r = e >> 6, d = e & 63;
    stage16(&sQ[e], &Qg[(qrow0 + r) * 512 + hoff + d]);
  }

  const v8f vzero = {};
  v8f o[4];
#pragma unroll
  for (int i = 0; i < 4; ++i) o[i] = vzero;
  float mrow[8], lrow[8];
#pragma unroll
  for (int r = 0; r < 8; ++r) { mrow[r] = -1e30f; lrow[r] = 0.0f; }

  unsigned short* Pw = &sP[wave * 512];

  for (int kt = 0; kt < Sk; kt += 32) {
    __syncthreads();
    // K tile: contiguous async copy [key][d].
#pragma unroll
    for (int t = 0; t < 2; ++t) {
      int e = (tid + t * 128) * 8;           // 2048 elements
      int r = e >> 6, d = e & 63;
      stage16(&sK[e], &Kg[(krow0 + kt + r) * 512 + hoff + d]);
    }
    // V tile: transpose to [d][key] (8 bf16 per 16B global read, ds scatter).
#pragma unroll
    for (int t = 0; t < 2; ++t) {
      int e = (tid + t * 128) * 8;
      int r = e >> 6, d0 = e & 63;
      uint4 vv = *(const uint4*)&Vg[(krow0 + kt + r) * 512 + hoff + d0];
      const unsigned short* pv = (const unsigned short*)&vv;
#pragma unroll
      for (int j = 0; j < 8; ++j) sV[(d0 + j) * 32 + r] = pv[j];
    }
    stage_wait();
    __syncthreads();

    // ---- S = Q @ K^T (16 q-rows x 32 keys per wave) ----
    v8f c0 = vzero, c1 = vzero;
    const int qrow = wave * 16 + ln;
#pragma unroll
    for (int s = 0; s < 2; ++s) {            // reduce d in two k=32 chunks
      v16bf a = ld_frag(&sQ[qrow * 64 + s * 32 + kh * 16],
                        &sQ[qrow * 64 + s * 32 + kh * 16 + 8]);
      v16bf b0 = ld_frag(&sK[(ln)      * 64 + s * 32 + kh * 16],
                         &sK[(ln)      * 64 + s * 32 + kh * 16 + 8]);
      c0 = WMMA_BF16(a, b0, c0);
      v16bf b1 = ld_frag(&sK[(16 + ln) * 64 + s * 32 + kh * 16],
                         &sK[(16 + ln) * 64 + s * 32 + kh * 16 + 8]);
      c1 = WMMA_BF16(a, b1, c1);
    }

    int mk0 = 1, mk1 = 1;
    if (masked) {
      mk0 = mask[krow0 + kt + ln];
      mk1 = mask[krow0 + kt + 16 + ln];
    }

    // ---- online softmax (row = r + 8*kh lives on 16 lanes) ----
#pragma unroll
    for (int r = 0; r < 8; ++r) {
      float s0 = c0[r] * 0.125f, s1 = c1[r] * 0.125f;   // 1/sqrt(64)
      if (masked) {
        if (!mk0) s0 = NEGF;
        if (!mk1) s1 = NEGF;
      }
      float mnew = fmaxf(mrow[r], rmax16(fmaxf(s0, s1)));
      float corr = __expf(mrow[r] - mnew);
      float p0 = __expf(s0 - mnew);
      float p1 = __expf(s1 - mnew);
      lrow[r] = lrow[r] * corr + rsum16(p0 + p1);
      mrow[r] = mnew;
#pragma unroll
      for (int jn = 0; jn < 4; ++jn) o[jn][r] *= corr;
      Pw[(r + 8 * kh) * 32 + ln]      = f2bf(p0);   // [row][key]
      Pw[(r + 8 * kh) * 32 + 16 + ln] = f2bf(p1);
    }

    // ---- O += P @ V ----
    {
      v16bf pa = ld_frag(&Pw[ln * 32 + kh * 16],
                         &Pw[ln * 32 + kh * 16 + 8]);
#pragma unroll
      for (int jn = 0; jn < 4; ++jn) {
        v16bf vb = ld_frag(&sV[(jn * 16 + ln) * 32 + kh * 16],
                           &sV[(jn * 16 + ln) * 32 + kh * 16 + 8]);
        o[jn] = WMMA_BF16(pa, vb, o[jn]);
      }
    }
  }

  // Normalize and store bf16: [b, q, h*64 + d]
#pragma unroll
  for (int jn = 0; jn < 4; ++jn) {
    const int d = jn * 16 + ln;
#pragma unroll
    for (int r = 0; r < 8; ++r) {
      int q = qt * 64 + wave * 16 + r + 8 * kh;
      Og[((size_t)b * Sq + q) * 512 + hoff + d] = f2bf(o[jn][r] * (1.0f / lrow[r]));
    }
  }
}

// ---------------------------------------------------------------------------
// LayerNorm over 256 features (f32), optional fused residual. Block per row.
// ---------------------------------------------------------------------------
__global__ __launch_bounds__(256) void ln_kernel(
    const float* __restrict__ a, const float* __restrict__ res,
    const float* __restrict__ g, const float* __restrict__ beta,
    float* __restrict__ out)
{
  __shared__ float sm[9];
  const int row = blockIdx.x;
  const int t = threadIdx.x;
  const size_t idx = (size_t)row * 256 + t;

  float v = a[idx];
  if (res) v += res[idx];

  float s = v;
#pragma unroll
  for (int m = 16; m >= 1; m >>= 1) s += __shfl_xor(s, m, 32);
  if ((t & 31) == 0) sm[t >> 5] = s;
  __syncthreads();
  if (t == 0) {
    float acc = 0.0f;
    for (int i = 0; i < 8; ++i) acc += sm[i];
    sm[8] = acc * (1.0f / 256.0f);
  }
  __syncthreads();
  const float mu = sm[8];
  __syncthreads();

  float d = v - mu;
  float q = d * d;
#pragma unroll
  for (int m = 16; m >= 1; m >>= 1) q += __shfl_xor(q, m, 32);
  if ((t & 31) == 0) sm[t >> 5] = q;
  __syncthreads();
  if (t == 0) {
    float acc = 0.0f;
    for (int i = 0; i < 8; ++i) acc += sm[i];
    sm[8] = acc * (1.0f / 256.0f);
  }
  __syncthreads();
  const float var = sm[8];

  out[idx] = d * rsqrtf(var + 1e-6f) * g[t] + beta[t];
}

// ---------------------------------------------------------------------------
extern "C" void kernel_launch(void* const* d_in, const int* in_sizes, int n_in,
                              void* d_out, int out_size, void* d_ws, size_t ws_size,
                              hipStream_t stream) {
  (void)in_sizes; (void)n_in; (void)out_size; (void)ws_size;

  const float* x        = (const float*)d_in[0];
  const float* enc      = (const float*)d_in[1];
  const int*   tgt_mask = (const int*)d_in[2];
  // src_mask (d_in[3]) unused: cross-attention is unmasked in the reference.
  const float* m1_wq = (const float*)d_in[4];  const float* m1_bq = (const float*)d_in[5];
  const float* m1_wk = (const float*)d_in[6];  const float* m1_bk = (const float*)d_in[7];
  const float* m1_wv = (const float*)d_in[8];  const float* m1_bv = (const float*)d_in[9];
  const float* m1_w2 = (const float*)d_in[10]; const float* m1_b2 = (const float*)d_in[11];
  const float* m1_g  = (const float*)d_in[12]; const float* m1_b  = (const float*)d_in[13];
  const float* m2_wq = (const float*)d_in[14]; const float* m2_bq = (const float*)d_in[15];
  const float* m2_wk = (const float*)d_in[16]; const float* m2_bk = (const float*)d_in[17];
  const float* m2_wv = (const float*)d_in[18]; const float* m2_bv = (const float*)d_in[19];
  const float* m2_w2 = (const float*)d_in[20]; const float* m2_b2 = (const float*)d_in[21];
  const float* m2_g  = (const float*)d_in[22]; const float* m2_b  = (const float*)d_in[23];
  const float* ln1_g = (const float*)d_in[24]; const float* ln1_b = (const float*)d_in[25];
  const float* ln2_g = (const float*)d_in[26]; const float* ln2_b = (const float*)d_in[27];
  const float* ln3_g = (const float*)d_in[28]; const float* ln3_b = (const float*)d_in[29];
  const float* ffn_w1 = (const float*)d_in[30]; const float* ffn_b1 = (const float*)d_in[31];
  const float* ffn_w2 = (const float*)d_in[32]; const float* ffn_b2 = (const float*)d_in[33];

  const int M = 8192;                 // B*S == B*T
  typedef unsigned short u16;

  char* p = (char*)d_ws;
  auto carve = [&](size_t bytes) -> char* {
    char* r = p; p += (bytes + 255) & ~(size_t)255; return r;
  };

  u16* xb   = (u16*)carve((size_t)M * 256 * 2);
  u16* encb = (u16*)carve((size_t)M * 256 * 2);
  u16* wT[10];
  for (int i = 0; i < 10; ++i) wT[i] = (u16*)carve((size_t)256 * 512 * 2);
  u16* Qb = (u16*)carve((size_t)M * 512 * 2);
  u16* Kb = (u16*)carve((size_t)M * 512 * 2);
  u16* Vb = (u16*)carve((size_t)M * 512 * 2);
  u16* Ob = (u16*)carve((size_t)M * 512 * 2);
  float* T1 = (float*)carve((size_t)M * 256 * 4);
  float* T2 = (float*)carve((size_t)M * 256 * 4);
  float* X1 = (float*)carve((size_t)M * 256 * 4);
  float* X2 = (float*)carve((size_t)M * 256 * 4);
  u16* X1b = (u16*)carve((size_t)M * 256 * 2);
  u16* X2b = (u16*)carve((size_t)M * 256 * 2);

  const int n4a = (M * 256) / 4;      // activation f32->bf16, 4/thread
  const int gCvtA = n4a / 256;
  const int nW = 256 * 512;
  const int gCvtW = nW / 256;

  dim3 gP(512 / 64, M / 64);          // GEMM N=512
  dim3 gO(256 / 64, M / 64);          // GEMM N=256
  dim3 gA(2048 / 64, 8, 4);           // attention: (q-tiles, H, B)

  // ---- one-time precision conversion / weight transposition ----
  cvt_kernel<<<gCvtA, 256, 0, stream>>>(x,   xb,   n4a);
  cvt_kernel<<<gCvtA, 256, 0, stream>>>(enc, encb, n4a);
  cvtT_kernel<<<gCvtW, 256, 0, stream>>>(m1_wq, wT[0], 256, 512);
  cvtT_kernel<<<gCvtW, 256, 0, stream>>>(m1_wk, wT[1], 256, 512);
  cvtT_kernel<<<gCvtW, 256, 0, stream>>>(m1_wv, wT[2], 256, 512);
  cvtT_kernel<<<gCvtW, 256, 0, stream>>>(m1_w2, wT[3], 512, 256);
  cvtT_kernel<<<gCvtW, 256, 0, stream>>>(m2_wq, wT[4], 256, 512);
  cvtT_kernel<<<gCvtW, 256, 0, stream>>>(m2_wk, wT[5], 256, 512);
  cvtT_kernel<<<gCvtW, 256, 0, stream>>>(m2_wv, wT[6], 256, 512);
  cvtT_kernel<<<gCvtW, 256, 0, stream>>>(m2_w2, wT[7], 512, 256);
  cvtT_kernel<<<gCvtW, 256, 0, stream>>>(ffn_w1, wT[8], 256, 512);
  cvtT_kernel<<<gCvtW, 256, 0, stream>>>(ffn_w2, wT[9], 512, 256);

  // ---- MHA1 (masked self-attention) ----
  gemm_bf16_kernel<<<gP, 128, 0, stream>>>(xb, wT[0], m1_bq, nullptr, Qb, M, 512, 256, 0, 1);
  gemm_bf16_kernel<<<gP, 128, 0, stream>>>(xb, wT[1], m1_bk, nullptr, Kb, M, 512, 256, 0, 1);
  gemm_bf16_kernel<<<gP, 128, 0, stream>>>(xb, wT[2], m1_bv, nullptr, Vb, M, 512, 256, 0, 1);
  attn_kernel<<<gA, 128, 0, stream>>>(Qb, Kb, Vb, tgt_mask, Ob, 2048, 2048, 1);
  gemm_bf16_kernel<<<gO, 128, 0, stream>>>(Ob, wT[3], m1_b2, x, T1, M, 256, 512, 0, 0);
  ln_kernel<<<M, 256, 0, stream>>>(T1, nullptr, m1_g, m1_b, T2);     // attn1
  ln_kernel<<<M, 256, 0, stream>>>(T2, x, ln1_g, ln1_b, X1);         // x = LN(x + attn1)
  cvt_kernel<<<gCvtA, 256, 0, stream>>>(X1, X1b, n4a);

  // ---- MHA2 (cross-attention, unmasked) ----
  gemm_bf16_kernel<<<gP, 128, 0, stream>>>(X1b,  wT[4], m2_bq, nullptr, Qb, M, 512, 256, 0, 1);
  gemm_bf16_kernel<<<gP, 128, 0, stream>>>(encb, wT[5], m2_bk, nullptr, Kb, M, 512, 256, 0, 1);
  gemm_bf16_kernel<<<gP, 128, 0, stream>>>(encb, wT[6], m2_bv, nullptr, Vb, M, 512, 256, 0, 1);
  attn_kernel<<<gA, 128, 0, stream>>>(Qb, Kb, Vb, nullptr, Ob, 2048, 2048, 0);
  gemm_bf16_kernel<<<gO, 128, 0, stream>>>(Ob, wT[7], m2_b2, X1, T1, M, 256, 512, 0, 0);
  ln_kernel<<<M, 256, 0, stream>>>(T1, nullptr, m2_g, m2_b, T2);     // attn2
  ln_kernel<<<M, 256, 0, stream>>>(T2, X1, ln2_g, ln2_b, X2);        // x = LN(x + attn2)
  cvt_kernel<<<gCvtA, 256, 0, stream>>>(X2, X2b, n4a);

  // ---- FFN ----
  gemm_bf16_kernel<<<gP, 128, 0, stream>>>(X2b, wT[8], ffn_b1, nullptr, Qb, M, 512, 256, 1, 1);
  gemm_bf16_kernel<<<gO, 128, 0, stream>>>(Qb,  wT[9], ffn_b2, X2, T1, M, 256, 512, 0, 0);
  ln_kernel<<<M, 256, 0, stream>>>(T1, nullptr, ln3_g, ln3_b, (float*)d_out);
}
